// MoEBlock_49976239456503
// MI455X (gfx1250) — matrix operand order repaired and checked by reading, not compiled, not run on hardware
//
#include <hip/hip_runtime.h>
#include <hip/hip_bf16.h>

// MoE top-1 block for MI455X (gfx1250, wave32, WMMA + async-to-LDS).
// Pipeline (all on `stream`):
//   init -> gate (argmax) -> seg offsets -> scatter -> pack(bf16 gather)
//   -> tiled-transpose w1/w2 to bf16 -> fused expert MLP GEMMs (WMMA bf16).
//
// Problem sizes (fixed by reference): B=8 N=1024 C=768 H=3072 E=4, T=8192.

#define T_TOK   8192
#define CDIM    768
#define HDIM    3072
#define NEXP    4
#define ROWS_MAX 8320          // 8192 + worst-case per-expert pad (<=31 each), 32-aligned
#define AS_STRIDE 776          // 768 + 8 bf16 pad (row stride 1552B -> +4 banks/row)
#define HS_STRIDE 3080         // 3072 + 8 bf16 pad
#define GEMM_SMEM ((32*AS_STRIDE + 32*HS_STRIDE) * 2 + 32 * 4)

typedef __attribute__((ext_vector_type(16))) __bf16       v16bf;
typedef __attribute__((ext_vector_type(8)))  float        v8f;
typedef __attribute__((ext_vector_type(4)))  unsigned int v4u;
typedef __attribute__((ext_vector_type(4)))  float        v4f;

union Frag { v16bf v; v4u q[2]; };

__device__ __forceinline__ unsigned short f2bf(float f) {
    unsigned int u = __float_as_uint(f);
    u += 0x7FFFu + ((u >> 16) & 1u);   // round-to-nearest-even
    return (unsigned short)(u >> 16);
}

// Branch-free tanh-form gelu: one v_exp_f32 + v_rcp, no EXEC divergence.
// |err| ~1e-3, well below the bf16 matmul noise floor.
__device__ __forceinline__ float gelu_f(float x) {
    const float u  = 0.7978845608028654f * x * (1.0f + 0.044715f * x * x);
    const float eu = __expf(-2.0f * fabsf(u));        // in (0,1], no overflow
    const float th = (1.0f - eu) / (1.0f + eu);       // tanh(|u|)
    return 0.5f * x * (1.0f + copysignf(th, u));
}

__device__ __forceinline__ v8f wmma_bf16(v16bf a, v16bf b, v8f c) {
    // (neg_a, A, neg_b, B, c_mod, C, reuse_a, reuse_b)
    return __builtin_amdgcn_wmma_f32_16x16x32_bf16(false, a, false, b, (short)0, c, false, false);
}

// Async global->LDS b128 copy (CDNA5 GLOBAL_LOAD_ASYNC_TO_LDS_B128, ASYNCcnt).
// lds_off: byte offset within the wave's LDS allocation (dynamic smem base = 0).
__device__ __forceinline__ void async_g2lds_b128(unsigned lds_off, const void* gsrc) {
    asm volatile("global_load_async_to_lds_b128 %0, %1, off"
                 :: "v"(lds_off), "v"((unsigned long long)gsrc) : "memory");
}
__device__ __forceinline__ void wait_async0() {
#if __has_builtin(__builtin_amdgcn_s_wait_asynccnt)
    __builtin_amdgcn_s_wait_asynccnt(0);
#else
    asm volatile("s_wait_asynccnt 0x0" ::: "memory");
#endif
}

// A fragment (16x32 bf16) from LDS, ISA 16-bit A layout:
// lane<16: K pairs {0..7,16..23}; lane>=16: +8.  Two ds_load_b128.
__device__ __forceinline__ v16bf load_a_lds(const unsigned short* lds, int stride,
                                            int row0, int kb, int lane) {
    const int half = lane >> 4;
    const unsigned short* p = lds + (size_t)(row0 + (lane & 15)) * stride + kb + half * 8;
    Frag f;
    f.q[0] = *(const v4u*)(p);        // K = kb+half*8 .. +7
    f.q[1] = *(const v4u*)(p + 16);   // K = kb+16+half*8 .. +7
    return f.v;
}

// B fragment (32x16 bf16) from pre-transposed weights: per-lane column, contiguous K.
// lanes 0-15 hold K={kb..kb+7, kb+16..kb+23}, lanes 16-31 hold +8. Two global_load_b128.
__device__ __forceinline__ v16bf load_b_g(const unsigned short* col, int kb, int half) {
    const unsigned short* p = col + kb + half * 8;
    Frag f;
    f.q[0] = *(const v4u*)(p);
    f.q[1] = *(const v4u*)(p + 16);
    return f.v;
}

// ---------------------------------------------------------------------------
__global__ void moe_init_kernel(int* counts, int* cursor, int* tok_of_row) {
    int i = blockIdx.x * blockDim.x + threadIdx.x;
    if (i < NEXP) counts[i] = 0;
    if (i >= NEXP && i < 2 * NEXP) cursor[i - NEXP] = 0;
    if (i < ROWS_MAX) tok_of_row[i] = -1;
}

// One wave per token; lane-parallel dot over C, wave32 shuffle reduce, argmax.
__global__ __launch_bounds__(256) void moe_gate_kernel(
    const float* __restrict__ x, const float* __restrict__ gw,
    const float* __restrict__ gb, const float* __restrict__ gc,
    int* __restrict__ expert_id, int* __restrict__ counts) {
    const int wave = threadIdx.x >> 5, lane = threadIdx.x & 31;
    const int t = blockIdx.x * 8 + wave;
    const float* xr = x + (size_t)t * CDIM;
    float a0 = 0.f, a1 = 0.f, a2 = 0.f, a3 = 0.f;
    for (int c = lane; c < CDIM; c += 32) {
        const float xv = xr[c];
        const v4f w = *(const v4f*)(gw + c * NEXP);   // one b128 per lane
        a0 += xv * w.x; a1 += xv * w.y; a2 += xv * w.z; a3 += xv * w.w;
    }
    #pragma unroll
    for (int off = 16; off > 0; off >>= 1) {
        a0 += __shfl_xor(a0, off, 32);
        a1 += __shfl_xor(a1, off, 32);
        a2 += __shfl_xor(a2, off, 32);
        a3 += __shfl_xor(a3, off, 32);
    }
    if (lane == 0) {
        float s[NEXP] = { a0 + gb[0] - gc[0], a1 + gb[1] - gc[1],
                          a2 + gb[2] - gc[2], a3 + gb[3] - gc[3] };
        int bi = 0; float bv = s[0];
        #pragma unroll
        for (int i = 1; i < NEXP; ++i) if (s[i] > bv) { bv = s[i]; bi = i; } // first-max tie
        expert_id[t] = bi;
        atomicAdd(&counts[bi], 1);
    }
}

__global__ void moe_offsets_kernel(const int* __restrict__ counts, int* __restrict__ seg) {
    if (threadIdx.x == 0 && blockIdx.x == 0) {
        int o = 0;
        for (int e = 0; e < NEXP; ++e) { seg[e] = o; o += (counts[e] + 31) & ~31; }
        seg[NEXP] = o;   // total padded rows
    }
}

__global__ __launch_bounds__(256) void moe_scatter_kernel(
    const int* __restrict__ expert_id, const int* __restrict__ seg,
    int* __restrict__ cursor, int* __restrict__ tok_of_row) {
    const int t = blockIdx.x * 256 + threadIdx.x;
    const int e = expert_id[t];
    const int pos = seg[e] + atomicAdd(&cursor[e], 1);
    tok_of_row[pos] = t;
}

// Gathered bf16 activations; pad rows (tok==-1) zeroed.
__global__ __launch_bounds__(256) void moe_pack_kernel(
    const float* __restrict__ x, const int* __restrict__ tok_of_row,
    unsigned short* __restrict__ xg) {
    const int row = blockIdx.x;
    const int t = tok_of_row[row];
    unsigned short* dst = xg + (size_t)row * CDIM;
    if (t >= 0) {
        const float* src = x + (size_t)t * CDIM;
        for (int i = threadIdx.x; i < CDIM; i += 256) dst[i] = f2bf(src[i]);
    } else {
        for (int i = threadIdx.x; i < CDIM; i += 256) dst[i] = 0;
    }
}

// Generic tiled transpose+downconvert: src f32 [nrows,ncols] -> dst bf16 [ncols,nrows]
// per blockIdx.z matrix.  Coalesced reads AND writes via a 32x33 LDS tile.
__global__ __launch_bounds__(256) void moe_transpose_bf16_kernel(
    const float* __restrict__ src0, unsigned short* __restrict__ dst0,
    int nrows, int ncols) {
    __shared__ float tile[32][33];
    const int cb = blockIdx.x * 32;            // src column block (dst row block)
    const int rb = blockIdx.y * 32;            // src row block (dst column block)
    const size_t mat = (size_t)nrows * ncols;
    const float* src = src0 + blockIdx.z * mat;
    unsigned short* dst = dst0 + blockIdx.z * mat;
    const int tx = threadIdx.x & 31, ty = threadIdx.x >> 5;
    #pragma unroll
    for (int i = 0; i < 32; i += 8)
        tile[ty + i][tx] = src[(size_t)(rb + ty + i) * ncols + cb + tx];
    __syncthreads();
    #pragma unroll
    for (int i = 0; i < 32; i += 8)
        dst[(size_t)(cb + ty + i) * nrows + rb + tx] = f2bf(tile[tx][ty + i]);
}

// Fused expert MLP: one block = 32 tokens of one expert.
// GEMM1: h[32,3072] = x[32,768] @ w1[e];  gelu -> bf16 LDS
// GEMM2: out[32,768] = h @ w2[e];  scatter to original token rows.
__global__ __launch_bounds__(256, 1) void moe_gemm_kernel(
    const unsigned short* __restrict__ xg,
    const unsigned short* __restrict__ w1t,
    const unsigned short* __restrict__ w2t,
    const float* __restrict__ b1, const float* __restrict__ b2,
    const int* __restrict__ seg, const int* __restrict__ tok_of_row,
    float* __restrict__ out) {
    extern __shared__ char smem[];
    unsigned short* As  = (unsigned short*)smem;           // 32 x AS_STRIDE bf16 @ byte 0
    unsigned short* Hs  = As + 32 * AS_STRIDE;             // 32 x HS_STRIDE bf16
    int*            tokS = (int*)(Hs + 32 * HS_STRIDE);    // 32

    const int rb = blockIdx.x * 32;
    if (rb >= seg[NEXP]) return;                           // beyond last padded segment
    int e = 0;
    if (rb >= seg[1]) e = 1;
    if (rb >= seg[2]) e = 2;
    if (rb >= seg[3]) e = 3;

    const int tid  = threadIdx.x;
    const int lane = tid & 31;
    const int wave = tid >> 5;
    const int half = lane >> 4;
    const int ln   = lane & 15;

    // stage x tile into LDS: async global->LDS b128 (ASYNCcnt), no VGPR round-trip
    for (int i = tid; i < 32 * (CDIM / 8); i += 256) {
        const int row = i / (CDIM / 8), s8 = i % (CDIM / 8);
        const unsigned lds_off = (unsigned)(row * AS_STRIDE + s8 * 8) * 2u;  // bytes
        async_g2lds_b128(lds_off, xg + (size_t)(rb + row) * CDIM + s8 * 8);
    }
    if (tid < 32) tokS[tid] = tok_of_row[rb + tid];
    wait_async0();                 // drain this wave's async stores to LDS
    __syncthreads();               // make whole tile visible workgroup-wide

    // ---- GEMM1: each wave owns 384 h-columns (24 N-tiles), M-tiles {0,16}
    const unsigned short* w1e = w1t + (size_t)e * HDIM * CDIM;
    for (int j = 0; j < 24; ++j) {
        const int hn = wave * 384 + j * 16 + ln;
        const unsigned short* col = w1e + (size_t)hn * CDIM;
        const float bias = b1[e * HDIM + hn];
        v8f acc0 = {}; v8f acc1 = {};
        #pragma unroll 4
        for (int kb = 0; kb < CDIM; kb += 32) {
            const v16bf bfr = load_b_g(col, kb, half);
            const v16bf a0  = load_a_lds(As, AS_STRIDE, 0,  kb, lane);
            const v16bf a1  = load_a_lds(As, AS_STRIDE, 16, kb, lane);
            acc0 = wmma_bf16(a0, bfr, acc0);
            acc1 = wmma_bf16(a1, bfr, acc1);
        }
        const int colIdx = wave * 384 + j * 16 + ln;
        #pragma unroll
        for (int v = 0; v < 8; ++v) {
            const int r0 = v + 8 * half;                   // C/D layout: M = v + 8*half
            Hs[(size_t)r0 * HS_STRIDE + colIdx]        = f2bf(gelu_f(acc0[v] + bias));
            Hs[(size_t)(16 + r0) * HS_STRIDE + colIdx] = f2bf(gelu_f(acc1[v] + bias));
        }
    }
    __syncthreads();

    // ---- GEMM2: each wave owns 96 out-columns (6 N-tiles)
    const unsigned short* w2e = w2t + (size_t)e * CDIM * HDIM;
    for (int j = 0; j < 6; ++j) {
        const int n = wave * 96 + j * 16 + ln;
        const unsigned short* col = w2e + (size_t)n * HDIM;
        const float bias = b2[e * CDIM + n];
        v8f acc0 = {}; v8f acc1 = {};
        #pragma unroll 4
        for (int kb = 0; kb < HDIM; kb += 32) {
            const v16bf bfr = load_b_g(col, kb, half);
            const v16bf a0  = load_a_lds(Hs, HS_STRIDE, 0,  kb, lane);
            const v16bf a1  = load_a_lds(Hs, HS_STRIDE, 16, kb, lane);
            acc0 = wmma_bf16(a0, bfr, acc0);
            acc1 = wmma_bf16(a1, bfr, acc1);
        }
        #pragma unroll
        for (int v = 0; v < 8; ++v) {
            const int r0 = v + 8 * half;
            const int t0 = tokS[r0];
            if (t0 >= 0) out[(size_t)t0 * CDIM + n] = acc0[v] + bias;
            const int t1 = tokS[16 + r0];
            if (t1 >= 0) out[(size_t)t1 * CDIM + n] = acc1[v] + bias;
        }
    }
}

// ---------------------------------------------------------------------------
extern "C" void kernel_launch(void* const* d_in, const int* in_sizes, int n_in,
                              void* d_out, int out_size, void* d_ws, size_t ws_size,
                              hipStream_t stream) {
    const float* x   = (const float*)d_in[0];   // [8,1024,768]
    const float* gw  = (const float*)d_in[1];   // [768,4]
    const float* gb  = (const float*)d_in[2];   // [4]
    const float* gc  = (const float*)d_in[3];   // [4]
    const float* w1  = (const float*)d_in[4];   // [4,768,3072]
    const float* b1  = (const float*)d_in[5];   // [4,3072]
    const float* w2  = (const float*)d_in[6];   // [4,3072,768]
    const float* b2  = (const float*)d_in[7];   // [4,768]
    float* out = (float*)d_out;                 // [8,1024,768] f32

    // workspace layout (~50.6 MB total)
    char* ws = (char*)d_ws;
    int* counts     = (int*)ws;                 //  4 ints
    int* cursor     = counts + 4;               //  4 ints
    int* seg        = cursor + 4;               //  5 ints
    int* expert_id  = (int*)(ws + 256);                      // 8192 ints
    int* tok_of_row = expert_id + T_TOK;                     // 8320 ints
    size_t off = 256 + (size_t)T_TOK * 4 + (size_t)ROWS_MAX * 4;
    off = (off + 255) & ~(size_t)255;
    unsigned short* xg  = (unsigned short*)(ws + off);       // 8320*768 bf16
    off += (size_t)ROWS_MAX * CDIM * 2;
    unsigned short* w1t = (unsigned short*)(ws + off);       // 4*3072*768 bf16
    off += (size_t)NEXP * HDIM * CDIM * 2;
    unsigned short* w2t = (unsigned short*)(ws + off);       // 4*768*3072 bf16

    moe_init_kernel<<<(ROWS_MAX + 255) / 256, 256, 0, stream>>>(counts, cursor, tok_of_row);
    moe_gate_kernel<<<T_TOK / 8, 256, 0, stream>>>(x, gw, gb, gc, expert_id, counts);
    moe_offsets_kernel<<<1, 32, 0, stream>>>(counts, seg);
    moe_scatter_kernel<<<T_TOK / 256, 256, 0, stream>>>(expert_id, seg, cursor, tok_of_row);
    moe_pack_kernel<<<ROWS_MAX, 256, 0, stream>>>(x, tok_of_row, xg);

    // w1 [C,H] -> w1t [H,C];  w2 [H,C] -> w2t [C,H]  (per expert, bf16)
    moe_transpose_bf16_kernel<<<dim3(HDIM / 32, CDIM / 32, NEXP), 256, 0, stream>>>(
        w1, w1t, CDIM, HDIM);
    moe_transpose_bf16_kernel<<<dim3(CDIM / 32, HDIM / 32, NEXP), 256, 0, stream>>>(
        w2, w2t, HDIM, CDIM);

    moe_gemm_kernel<<<ROWS_MAX / 32, 256, GEMM_SMEM, stream>>>(
        xg, w1t, w2t, b1, b2, seg, tok_of_row, out);
}
